// PhaseBoundaryPenalty_70265664962707
// MI455X (gfx1250) — matrix operand order repaired and checked by reading, not compiled, not run on hardware
//
#include <hip/hip_runtime.h>
#include <hip/hip_bf16.h>
#include <math.h>

// ---------------------------------------------------------------------------
// SRSF/DTW entropy pipeline for gfx1250 (MI455X).
//   F = tanh(C@W1+b1)@W2 + b2  -> WMMA f16->f32 GEMM
//   per sample: SRSF, 4x DTW (anti-diagonal wavefront DP in LDS),
//               backtrack -> gamma -> warp -> SRSF -> trapz dist, entropy.
// ---------------------------------------------------------------------------

typedef __attribute__((ext_vector_type(16))) _Float16 v16h;
typedef __attribute__((ext_vector_type(8)))  _Float16 v8h;
typedef __attribute__((ext_vector_type(8)))  float    v8f;

#define NS   1024      // samples
#define ND   256       // domain
#define NH   512       // hidden
#define NT   4         // templates
#define BIGF 1e9f

// ---------------- stage 1: H = tanh(C@W1 + b1) as f16 ----------------------
__global__ void srsf_hidden_f16(const float* __restrict__ X,
                                const float* __restrict__ W1,
                                const float* __restrict__ b1,
                                _Float16* __restrict__ H) {
  int idx = blockIdx.x * blockDim.x + threadIdx.x;
  if (idx >= NS * NH) return;
  int s = idx >> 9;          // /512
  int h = idx & (NH - 1);
  float v = X[s * 2 + 0] * W1[h] + X[s * 2 + 1] * W1[NH + h] + b1[h];
  H[idx] = (_Float16)tanhf(v);
}

// ---------------- stage 1b: W2 [512,256] -> W2T [256,512] f16 --------------
__global__ void srsf_w2t_f16(const float* __restrict__ W2,
                             _Float16* __restrict__ W2T) {
  int idx = blockIdx.x * blockDim.x + threadIdx.x;
  if (idx >= NH * ND) return;
  int k = idx >> 8;          // /256
  int n = idx & (ND - 1);
  W2T[n * NH + k] = (_Float16)W2[k * ND + n];
}

// ---------------- stage 2: F = H @ W2 + b2 via WMMA ------------------------
// One wave computes one 16x16 tile of F. K=512 -> 16 chained WMMAs.
// A (16x32 f16) lane layout: row m = lane&15; hs = lane>>4:
//   elems 0..7 -> K = 8*hs + 0..7 ; elems 8..15 -> K = 16 + 8*hs + 0..7
// B (32x16 f16) lane layout: col n = lane&15; same K mapping as A
// C/D (16x16 f32): vgpr v -> row m = v + 8*hs, col n = lane&15
__global__ void srsf_gemm_wmma(const _Float16* __restrict__ H,
                               const _Float16* __restrict__ W2T,
                               const float* __restrict__ b2,
                               float* __restrict__ F) {
  const int wave = threadIdx.x >> 5;
  const int lane = threadIdx.x & 31;
  const int tile = blockIdx.x * (blockDim.x >> 5) + wave;  // 0..1023
  const int mt = tile >> 4;                                // 0..63
  const int nt = tile & 15;                                // 0..15
  const int mb = mt * 16, nb = nt * 16;
  const int hs = lane >> 4;
  const int ln = lane & 15;
  const int m  = mb + ln;                                  // A row for lane
  const int n  = nb + ln;                                  // B/D col for lane

  v8f acc;
  const float bias = b2[n];
#pragma unroll
  for (int v = 0; v < 8; ++v) acc[v] = bias;

  const _Float16* __restrict__ arow = H   + m * NH;
  const _Float16* __restrict__ bcol = W2T + n * NH;

  for (int kb = 0; kb < NH; kb += 32) {
    v8h a0 = *(const v8h*)(arow + kb + 8 * hs);
    v8h a1 = *(const v8h*)(arow + kb + 16 + 8 * hs);
    v8h b0 = *(const v8h*)(bcol + kb + 8 * hs);
    v8h b1 = *(const v8h*)(bcol + kb + 16 + 8 * hs);
    v16h a, b;
#pragma unroll
    for (int e = 0; e < 8; ++e) {
      a[e] = a0[e]; a[8 + e] = a1[e];
      b[e] = b0[e]; b[8 + e] = b1[e];
    }
    acc = __builtin_amdgcn_wmma_f32_16x16x32_f16(
        /*neg_a=*/false, a, /*neg_b=*/false, b,
        /*c_mod=*/(short)0, acc, /*reuse_a=*/false, /*reuse_b=*/false);
  }
#pragma unroll
  for (int v = 0; v < 8; ++v)
    F[(mb + v + 8 * hs) * ND + n] = acc[v];
}

// ---------------- stage 3: per-sample DTW + entropy ------------------------
// 1 block (256 threads = 8 waves) per sample. Wavefront DP, moves packed
// 2 bits/cell in thread-private LDS rows (padded to 132B to avoid bank
// conflicts on the per-diagonal byte RMW).
__global__ void __launch_bounds__(256)
srsf_dtw_entropy(const float* __restrict__ F_all,
                 const float* __restrict__ Tq,
                 float* __restrict__ out) {
  __shared__ float sF[ND];
  __shared__ float sq[ND];
  __shared__ float sqt[ND];
  __shared__ float dp[3][ND];
  __shared__ float sgam[ND];
  __shared__ float red[ND];
  __shared__ float sd2[NT];
  __shared__ unsigned char mv[ND][132];   // 2-bit codes, 4 per byte, by diag d

  const int s = blockIdx.x;
  const int i = threadIdx.x;
  const float dt = 1.0f / 255.0f;
  const float inv_dt = 255.0f;

  sF[i] = F_all[s * ND + i];
  __syncthreads();
  {  // q = SRSF(F): jnp.gradient (central, one-sided ends) / dt
    float g;
    if (i == 0)            g = (sF[1] - sF[0]) * inv_dt;
    else if (i == ND - 1)  g = (sF[ND - 1] - sF[ND - 2]) * inv_dt;
    else                   g = (sF[i + 1] - sF[i - 1]) * (0.5f * inv_dt);
    sq[i] = g / sqrtf(fabsf(g) + 1e-8f);
  }
  __syncthreads();

  for (int tq = 0; tq < NT; ++tq) {
    sqt[i] = Tq[tq * ND + i];
    __syncthreads();
    const float qt_i = sqt[i];

    // DP init: d=0 diag (only cell (0,0) valid), d=-1 diag = BIG
    dp[2][i] = BIGF;
    float e0 = qt_i - sq[0];
    dp[0][i] = (i == 0) ? e0 * e0 : BIGF;
    mv[i][0] = 0;   // move code for d=0 is 0 (diag)
    __syncthreads();

    for (int d = 1; d <= 2 * ND - 2; ++d) {
      float*       cur   = dp[d % 3];
      const float* prev  = dp[(d + 2) % 3];   // d-1
      const float* prev2 = dp[(d + 1) % 3];   // d-2
      const int j = d - i;
      const bool valid = (j >= 0) && (j < ND);
      const float qj = sq[valid ? j : 0];
      const float c  = valid ? (qt_i - qj) * (qt_i - qj) : BIGF;
      const float dg = (i > 0) ? prev2[i - 1] : BIGF;  // (i-1, j-1)
      const float up = (i > 0) ? prev[i - 1]  : BIGF;  // (i-1, j)
      const float lf = (d - i - 1 >= 0) ? prev[i] : BIGF;  // (i, j-1)
      int   m    = 0;
      float best = dg;                      // argmin order: diag, up, left
      if (up < best) { best = up; m = 1; }
      if (lf < best) { best = lf; m = 2; }
      cur[i] = valid ? (c + best) : BIGF;
      const int bidx = d >> 2, sh = (d & 3) * 2;
      unsigned char byte = (sh == 0) ? (unsigned char)0 : mv[i][bidx];
      mv[i][bidx] = (unsigned char)(byte | (m << sh));
      __syncthreads();
    }

    // backtrack + gamma scatter-max + cummax (thread 0, reads LDS only)
    sgam[i] = 0.0f;
    __syncthreads();
    if (i == 0) {
      int pi = ND - 1, pj = ND - 1;
      sgam[pi] = fmaxf(sgam[pi], (float)pj * dt);
      for (int step = 0; step < 2 * ND - 2; ++step) {
        const int d = pi + pj;
        const int m = (mv[pi][d >> 2] >> ((d & 3) * 2)) & 3;
        if (!(pi == 0 && pj == 0)) {
          pi -= (m != 2);
          pj -= (m != 1);
        }
        sgam[pi] = fmaxf(sgam[pi], (float)pj * dt);
      }
      float run = sgam[0];
      for (int k = 1; k < ND; ++k) { run = fmaxf(run, sgam[k]); sgam[k] = run; }
    }
    __syncthreads();

    // f_gam = interp(gamma, t, F)  (uniform t on [0,1])
    {
      float pos = sgam[i] * 255.0f;
      int k0 = (int)floorf(pos);
      k0 = k0 < 0 ? 0 : (k0 > ND - 2 ? ND - 2 : k0);
      float fr = pos - (float)k0;
      dp[0][i] = sF[k0] + (sF[k0 + 1] - sF[k0]) * fr;
    }
    __syncthreads();

    // q_gam = SRSF(f_gam); integrand (qt - q_gam)^2 with trapezoid weights
    {
      float g;
      if (i == 0)           g = (dp[0][1] - dp[0][0]) * inv_dt;
      else if (i == ND - 1) g = (dp[0][ND - 1] - dp[0][ND - 2]) * inv_dt;
      else                  g = (dp[0][i + 1] - dp[0][i - 1]) * (0.5f * inv_dt);
      float qg = g / sqrtf(fabsf(g) + 1e-8f);
      float y  = (qt_i - qg) * (qt_i - qg);
      red[i] = (i == 0 || i == ND - 1) ? 0.5f * y : y;
    }
    __syncthreads();
    for (int off = 128; off > 0; off >>= 1) {
      if (i < off) red[i] += red[i + off];
      __syncthreads();
    }
    if (i == 0) sd2[tq] = dt * red[0] + 1e-12f;   // dist^2 = trapz + 1e-12
    __syncthreads();
  }

  if (i == 0) {
    float d2[NT], mean = 0.0f;
#pragma unroll
    for (int k = 0; k < NT; ++k) { d2[k] = sd2[k]; mean += d2[k]; }
    const float eps = mean * (1.0f / NT) + 1e-8f;
    float l[NT], mx = -BIGF;
#pragma unroll
    for (int k = 0; k < NT; ++k) { l[k] = -d2[k] / eps; mx = fmaxf(mx, l[k]); }
    float sum = 0.0f;
#pragma unroll
    for (int k = 0; k < NT; ++k) { l[k] = expf(l[k] - mx); sum += l[k]; }
    float ent = 0.0f;
#pragma unroll
    for (int k = 0; k < NT; ++k) {
      float p = l[k] / sum;
      ent -= p * logf(p + 1e-12f);
    }
    out[s] = ent;
  }
}

// ---------------------------------------------------------------------------
extern "C" void kernel_launch(void* const* d_in, const int* in_sizes, int n_in,
                              void* d_out, int out_size, void* d_ws, size_t ws_size,
                              hipStream_t stream) {
  const float* X  = (const float*)d_in[0];   // [1,1024,2]
  // d_in[1] = t (uniform linspace; dt hardcoded as 1/255)
  const float* W1 = (const float*)d_in[2];   // [2,512]
  const float* b1 = (const float*)d_in[3];   // [512]
  const float* W2 = (const float*)d_in[4];   // [512,256]
  const float* b2 = (const float*)d_in[5];   // [256]
  const float* Tq = (const float*)d_in[6];   // [4,256]
  float* out = (float*)d_out;                // [1024]

  char* ws = (char*)d_ws;
  _Float16* H   = (_Float16*)ws;                                // 1,048,576 B
  _Float16* W2T = (_Float16*)(ws + (size_t)NS * NH * 2);        //   262,144 B
  float*    F   = (float*)(ws + (size_t)NS * NH * 2 + (size_t)ND * NH * 2);

  srsf_hidden_f16<<<(NS * NH + 255) / 256, 256, 0, stream>>>(X, W1, b1, H);
  srsf_w2t_f16<<<(NH * ND + 255) / 256, 256, 0, stream>>>(W2, W2T);
  // 1024 tiles (64 M-tiles x 16 N-tiles), 4 waves/block -> 256 blocks
  srsf_gemm_wmma<<<256, 128, 0, stream>>>(H, W2T, b2, F);
  srsf_dtw_entropy<<<NS, ND, 0, stream>>>(F, Tq, out);
}
